// CostVolume_41858751267256
// MI455X (gfx1250) — matrix operand order repaired and checked by reading, not compiled, not run on hardware
//
#include <hip/hip_runtime.h>

typedef __attribute__((ext_vector_type(16))) __bf16 v16bf;
typedef __attribute__((ext_vector_type(8)))  float  v8f;

union U16bf { v16bf v; uint4 q[2]; };

// ---------------------------------------------------------------------------
// Weight reorder: src is f32 [K, 64] (K = Stotal*32), dst is bf16 in WMMA
// B-matrix order: dst[((S*2+half)*64 + n)*16 + e] = src[(S*32 + half*8 + (e&7) + 16*(e>>3))*64 + n]
// so a lane (n, half) reads 16 contiguous bf16 (= 2 x b128) per 32-wide k-step.
// ---------------------------------------------------------------------------
__global__ void k_reorder_w(const float* __restrict__ src, __bf16* __restrict__ dst, int Stotal) {
  int N = Stotal * 2048;
  for (int i = blockIdx.x * blockDim.x + threadIdx.x; i < N; i += gridDim.x * blockDim.x) {
    int e  = i & 15;
    int n  = (i >> 4) & 63;
    int hf = (i >> 10) & 1;
    int S  = i >> 11;
    int k  = S * 32 + hf * 8 + (e & 7) + 16 * (e >> 3);
    dst[i] = (__bf16)src[k * 64 + n];
  }
}

__device__ __forceinline__ void load_bfrag(const __bf16* bp, U16bf& bm) {
  bm.q[0] = *(const uint4*)bp;
  bm.q[1] = *(const uint4*)(bp + 8);
}

__device__ __forceinline__ void load_bquad(const __bf16* base, int s, int hf, int r,
                                           U16bf* dst) {
  const __bf16* bp = base + ((s * 2 + hf) * 64 + r) * 16;
#pragma unroll
  for (int j = 0; j < 4; ++j) load_bfrag(bp + j * 256, dst[j]);
}

// ---------------------------------------------------------------------------
// Downsample: y[p,n] = relu(BN(sum_c x[p,c]*W[c,n] + b)), p in [0,7680), bf16 out.
// One wave per 16x64 tile (A fragment reused across 4 N-tiles); K=256 -> 8 steps.
// ---------------------------------------------------------------------------
__global__ void k_downsample(const float* __restrict__ L, const float* __restrict__ R,
                             const __bf16* __restrict__ dsr,
                             const float* __restrict__ cbias,
                             const float* __restrict__ bng, const float* __restrict__ bnb,
                             const float* __restrict__ bnm, const float* __restrict__ bnv,
                             __bf16* __restrict__ lf, __bf16* __restrict__ rf) {
  int wave = threadIdx.x >> 5;
  int lane = threadIdx.x & 31;
  int t = blockIdx.x * (blockDim.x >> 5) + wave;     // [0, 960)
  int ptile = t % 480;
  int img   = t / 480;
  const float* in  = img ? R : L;
  __bf16*      out = img ? rf : lf;

  int hf = lane >> 4;
  int r  = lane & 15;
  int p  = ptile * 16 + r;                           // A row (position)

  float scale[4], shift[4];
#pragma unroll
  for (int j = 0; j < 4; ++j) {
    int n = j * 16 + r;
    scale[j] = bng[n] * rsqrtf(bnv[n] + 1e-3f);
    shift[j] = (cbias[n] - bnm[n]) * scale[j] + bnb[n];
  }

  v8f acc[4] = {};
#pragma unroll
  for (int ks = 0; ks < 8; ++ks) {
    const float* ap = in + p * 256 + ks * 32 + hf * 8;
    v16bf av;
#pragma unroll
    for (int j = 0; j < 8; ++j) {
      av[j]     = (__bf16)ap[j];
      av[8 + j] = (__bf16)ap[16 + j];
    }
#pragma unroll
    for (int j = 0; j < 4; ++j) {
      U16bf bm;
      load_bfrag(dsr + ((ks * 2 + hf) * 64 + j * 16 + r) * 16, bm);
      acc[j] = __builtin_amdgcn_wmma_f32_16x16x32_bf16(false, av, false, bm.v,
                                                       (short)0, acc[j], false, false);
    }
  }
#pragma unroll
  for (int j = 0; j < 4; ++j) {
#pragma unroll
    for (int vv = 0; vv < 8; ++vv) {
      int po = ptile * 16 + vv + 8 * hf;
      float y = acc[j][vv] * scale[j] + shift[j];
      out[po * 64 + j * 16 + r] = (__bf16)fmaxf(y, 0.0f);
    }
  }
}

// ---------------------------------------------------------------------------
// Conv1: implicit GEMM over the cost volume built on the fly from lf/rf.
// Volume dims (z,y,x) = (h=48, w=160, d=48), C_in = 128 (64 left | 64 right).
// One wave owns a 16x64 output tile. Steps s = tap*4 + ks (108 total), with the
// B fragments software-pipelined one step ahead in a register double buffer.
// ---------------------------------------------------------------------------
__global__ void k_conv1(const __bf16* __restrict__ lf, const __bf16* __restrict__ rf,
                        const __bf16* __restrict__ w1r,
                        const float* __restrict__ cbias,
                        const float* __restrict__ bng, const float* __restrict__ bnb,
                        const float* __restrict__ bnm, const float* __restrict__ bnv,
                        __bf16* __restrict__ mid) {
  int wave = threadIdx.x >> 5;
  int lane = threadIdx.x & 31;
  int ptile = blockIdx.x * (blockDim.x >> 5) + wave; // [0, 23040)

  int hf = lane >> 4;
  int r  = lane & 15;
  int pos = ptile * 16 + r;
  int d  = pos % 48;
  int w_ = (pos / 48) % 160;
  int h  = pos / 7680;

  float scale[4], shift[4];
#pragma unroll
  for (int j = 0; j < 4; ++j) {
    int n = j * 16 + r;
    scale[j] = bng[n] * rsqrtf(bnv[n] + 1e-3f);
    shift[j] = (cbias[n] - bnm[n]) * scale[j] + bnb[n];
  }

  v8f acc[4] = {};
  uint4 z4 = make_uint4(0u, 0u, 0u, 0u);

  auto stepf = [&](int s, U16bf* bcur, U16bf* bnext) {
    int tap = s >> 2, ks = s & 3;
    int dz = tap / 9 - 1, dy = (tap / 3) % 3 - 1, dx = tap % 3 - 1;
    int hh = h + dz, ww = w_ + dy, dd = d + dx;
    bool inb = ((unsigned)hh < 48u) && ((unsigned)ww < 160u) && ((unsigned)dd < 48u);
    int  wd  = ww - dd;
    bool valid = inb && (wd >= 0);
    U16bf a;
    if (valid) {
      const __bf16* ap = (ks < 2 ? lf + (hh * 160 + ww) * 64 + ks * 32
                                 : rf + (hh * 160 + wd) * 64 + (ks - 2) * 32) + hf * 8;
      a.q[0] = *(const uint4*)ap;
      a.q[1] = *(const uint4*)(ap + 16);
    } else {
      a.q[0] = z4; a.q[1] = z4;
    }
    if (s + 1 < 108) load_bquad(w1r, s + 1, hf, r, bnext);
    if ((s & 3) == 0 && s + 8 < 108)
      __builtin_prefetch(w1r + (((s + 8) * 2 + hf) * 64 + r) * 16, 0, 1);
#pragma unroll
    for (int j = 0; j < 4; ++j)
      acc[j] = __builtin_amdgcn_wmma_f32_16x16x32_bf16(false, a.v, false, bcur[j].v,
                                                       (short)0, acc[j], false, false);
  };

  U16bf b0[4], b1[4];
  load_bquad(w1r, 0, hf, r, b0);
  for (int s = 0; s < 108; s += 2) {
    stepf(s,     b0, b1);
    stepf(s + 1, b1, b0);
  }

#pragma unroll
  for (int j = 0; j < 4; ++j) {
#pragma unroll
    for (int vv = 0; vv < 8; ++vv) {
      int po = ptile * 16 + vv + 8 * hf;
      float y = acc[j][vv] * scale[j] + shift[j];
      mid[po * 64 + j * 16 + r] = (__bf16)fmaxf(y, 0.0f);
    }
  }
}

// ---------------------------------------------------------------------------
// Conv2: 64 -> 64 over the dense mid volume, steps s = tap*2 + ks (54 total),
// B double-buffered one step ahead. Writes f32 (BN+ReLU fused) to d_out.
// ---------------------------------------------------------------------------
__global__ void k_conv2(const __bf16* __restrict__ mid, const __bf16* __restrict__ w2r,
                        const float* __restrict__ cbias,
                        const float* __restrict__ bng, const float* __restrict__ bnb,
                        const float* __restrict__ bnm, const float* __restrict__ bnv,
                        float* __restrict__ out) {
  int wave = threadIdx.x >> 5;
  int lane = threadIdx.x & 31;
  int ptile = blockIdx.x * (blockDim.x >> 5) + wave;

  int hf = lane >> 4;
  int r  = lane & 15;
  int pos = ptile * 16 + r;
  int d  = pos % 48;
  int w_ = (pos / 48) % 160;
  int h  = pos / 7680;

  float scale[4], shift[4];
#pragma unroll
  for (int j = 0; j < 4; ++j) {
    int n = j * 16 + r;
    scale[j] = bng[n] * rsqrtf(bnv[n] + 1e-3f);
    shift[j] = (cbias[n] - bnm[n]) * scale[j] + bnb[n];
  }

  v8f acc[4] = {};
  uint4 z4 = make_uint4(0u, 0u, 0u, 0u);

  auto stepf = [&](int s, U16bf* bcur, U16bf* bnext) {
    int tap = s >> 1, ks = s & 1;
    int dz = tap / 9 - 1, dy = (tap / 3) % 3 - 1, dx = tap % 3 - 1;
    int hh = h + dz, ww = w_ + dy, dd = d + dx;
    bool inb = ((unsigned)hh < 48u) && ((unsigned)ww < 160u) && ((unsigned)dd < 48u);
    U16bf a;
    if (inb) {
      const __bf16* ap = mid + ((hh * 160 + ww) * 48 + dd) * 64 + ks * 32 + hf * 8;
      a.q[0] = *(const uint4*)ap;
      a.q[1] = *(const uint4*)(ap + 16);
    } else {
      a.q[0] = z4; a.q[1] = z4;
    }
    if (s + 1 < 54) load_bquad(w2r, s + 1, hf, r, bnext);
    if ((s & 1) == 0 && s + 4 < 54)
      __builtin_prefetch(w2r + (((s + 4) * 2 + hf) * 64 + r) * 16, 0, 1);
#pragma unroll
    for (int j = 0; j < 4; ++j)
      acc[j] = __builtin_amdgcn_wmma_f32_16x16x32_bf16(false, a.v, false, bcur[j].v,
                                                       (short)0, acc[j], false, false);
  };

  U16bf b0[4], b1[4];
  load_bquad(w2r, 0, hf, r, b0);
  for (int s = 0; s < 54; s += 2) {
    stepf(s,     b0, b1);
    stepf(s + 1, b1, b0);
  }

#pragma unroll
  for (int j = 0; j < 4; ++j) {
#pragma unroll
    for (int vv = 0; vv < 8; ++vv) {
      int po = ptile * 16 + vv + 8 * hf;
      float y = acc[j][vv] * scale[j] + shift[j];
      out[po * 64 + j * 16 + r] = fmaxf(y, 0.0f);
    }
  }
}

// ---------------------------------------------------------------------------
extern "C" void kernel_launch(void* const* d_in, const int* in_sizes, int n_in,
                              void* d_out, int out_size, void* d_ws, size_t ws_size,
                              hipStream_t stream) {
  (void)in_sizes; (void)n_in; (void)out_size; (void)ws_size;

  const float* L     = (const float*)d_in[0];
  const float* R     = (const float*)d_in[1];
  const float* ds_w  = (const float*)d_in[2];
  const float* ds_b  = (const float*)d_in[3];
  const float* bn0_g = (const float*)d_in[4];
  const float* bn0_b = (const float*)d_in[5];
  const float* bn0_m = (const float*)d_in[6];
  const float* bn0_v = (const float*)d_in[7];
  const float* c1_w  = (const float*)d_in[8];
  const float* c1_b  = (const float*)d_in[9];
  const float* bn1_g = (const float*)d_in[10];
  const float* bn1_b = (const float*)d_in[11];
  const float* bn1_m = (const float*)d_in[12];
  const float* bn1_v = (const float*)d_in[13];
  const float* c2_w  = (const float*)d_in[14];
  const float* c2_b  = (const float*)d_in[15];
  const float* bn2_g = (const float*)d_in[16];
  const float* bn2_b = (const float*)d_in[17];
  const float* bn2_m = (const float*)d_in[18];
  const float* bn2_v = (const float*)d_in[19];

  char* ws = (char*)d_ws;
  __bf16* lf  = (__bf16*)(ws);                 //  983040 B
  __bf16* rf  = (__bf16*)(ws + 983040);        //  983040 B
  __bf16* mid = (__bf16*)(ws + 1966080);       // 47185920 B
  __bf16* dsr = (__bf16*)(ws + 49152000);      //   32768 B (8*2048 bf16)
  __bf16* w1r = (__bf16*)(ws + 49168384);      //  442368 B (108*2048 bf16)
  __bf16* w2r = (__bf16*)(ws + 49610752);      //  221184 B (54*2048 bf16)

  // Weight reorders into WMMA B-operand layout (bf16).
  k_reorder_w<<<(8   * 2048 + 255) / 256, 256, 0, stream>>>(ds_w, dsr, 8);
  k_reorder_w<<<(108 * 2048 + 255) / 256, 256, 0, stream>>>(c1_w, w1r, 108);
  k_reorder_w<<<(54  * 2048 + 255) / 256, 256, 0, stream>>>(c2_w, w2r, 54);

  // Downsample both images: 2 * 480 waves (16x64 tile per wave), 4 waves/block.
  k_downsample<<<240, 128, 0, stream>>>(L, R, dsr, ds_b,
                                        bn0_g, bn0_b, bn0_m, bn0_v, lf, rf);

  // Conv1 over implicit cost volume: 23040 waves (16x64 tile), 4 waves/block.
  k_conv1<<<5760, 128, 0, stream>>>(lf, rf, w1r, c1_b,
                                    bn1_g, bn1_b, bn1_m, bn1_v, mid);

  // Conv2 -> f32 output.
  k_conv2<<<5760, 128, 0, stream>>>(mid, w2r, c2_b,
                                    bn2_g, bn2_b, bn2_m, bn2_v, (float*)d_out);
}
